// PseudoLogLSTM_13460427506024
// MI455X (gfx1250) — matrix operand order repaired and checked by reading, not compiled, not run on hardware
//
#include <hip/hip_runtime.h>
#include <hip/hip_bf16.h>

// ---------------------------------------------------------------------------
// PseudoLogLSTM on MI455X (gfx1250, wave32, WMMA)
//
// gates[128,2048] = concat(x_t, h_{t-1}+h_{t-2})[128,1024] @ WcombT[1024,2048]
// computed with v_wmma_f32_16x16x32_f16 (f16 operands, f32 accum).
// Persistent kernel: 32 WGs x 256 threads, one grid barrier per layer-step.
// WG w owns h-columns [16w,16w+16) of all 4 gate blocks -> LSTM cell done
// entirely in registers (i,f,g,o tiles live in the same wave).
// NEW: weight slabs staged into LDS with GLOBAL_LOAD_ASYNC_TO_LDS_B128
// (ASYNCcnt path, no VGPR round-trip), synced with s_wait_asynccnt.
// ---------------------------------------------------------------------------

#define T_  512
#define B_  128
#define H_  512
#define G_  2048           // 4*H
#define K_  1024           // 2*H (x half + h half)
#define NWG 32
#define NK  32             // K_/32 k-steps
#define LDS_PITCH 72       // ushorts per LDS B-row (conflict-free b128 reads)

typedef __attribute__((ext_vector_type(16))) _Float16 v16h;
typedef __attribute__((ext_vector_type(8)))  float    v8f;

// ---------------- math helpers ----------------
__device__ __forceinline__ float fsig(float x)  { return 1.0f / (1.0f + __expf(-x)); }
__device__ __forceinline__ float ftanh(float x) { return 1.0f - 2.0f / (1.0f + __expf(2.0f * x)); }

__device__ __forceinline__ v8f wmma16(v16h a, v16h b, v8f c) {
  // D = A(16x32 f16) * B(32x16 f16) + C(16x16 f32)
  return __builtin_amdgcn_wmma_f32_16x16x32_f16(false, a, false, b, (short)0, c,
                                                false, false);
}

// ---------------- A-fragment builders (16-bit A 16x32 layout) ----------------
// lane L: M = L&15, kh = L>>4; frag elems 0..7 = K kh*8..kh*8+7,
//                              frag elems 8..15 = K 16+kh*8..16+kh*8+7
__device__ __forceinline__ v16h buildA1(const float* __restrict__ p) {
  const float4 u0 = *(const float4*)(p);
  const float4 u1 = *(const float4*)(p + 4);
  const float4 u2 = *(const float4*)(p + 16);
  const float4 u3 = *(const float4*)(p + 20);
  v16h a;
  a[0]=(_Float16)u0.x;  a[1]=(_Float16)u0.y;  a[2]=(_Float16)u0.z;  a[3]=(_Float16)u0.w;
  a[4]=(_Float16)u1.x;  a[5]=(_Float16)u1.y;  a[6]=(_Float16)u1.z;  a[7]=(_Float16)u1.w;
  a[8]=(_Float16)u2.x;  a[9]=(_Float16)u2.y;  a[10]=(_Float16)u2.z; a[11]=(_Float16)u2.w;
  a[12]=(_Float16)u3.x; a[13]=(_Float16)u3.y; a[14]=(_Float16)u3.z; a[15]=(_Float16)u3.w;
  return a;
}

__device__ __forceinline__ v16h buildA2(const float* __restrict__ p,
                                        const float* __restrict__ q) {
  const float4 a0 = *(const float4*)(p);      const float4 b0 = *(const float4*)(q);
  const float4 a1 = *(const float4*)(p + 4);  const float4 b1 = *(const float4*)(q + 4);
  const float4 a2 = *(const float4*)(p + 16); const float4 b2 = *(const float4*)(q + 16);
  const float4 a3 = *(const float4*)(p + 20); const float4 b3 = *(const float4*)(q + 20);
  v16h a;
  a[0]=(_Float16)(a0.x+b0.x);  a[1]=(_Float16)(a0.y+b0.y);
  a[2]=(_Float16)(a0.z+b0.z);  a[3]=(_Float16)(a0.w+b0.w);
  a[4]=(_Float16)(a1.x+b1.x);  a[5]=(_Float16)(a1.y+b1.y);
  a[6]=(_Float16)(a1.z+b1.z);  a[7]=(_Float16)(a1.w+b1.w);
  a[8]=(_Float16)(a2.x+b2.x);  a[9]=(_Float16)(a2.y+b2.y);
  a[10]=(_Float16)(a2.z+b2.z); a[11]=(_Float16)(a2.w+b2.w);
  a[12]=(_Float16)(a3.x+b3.x); a[13]=(_Float16)(a3.y+b3.y);
  a[14]=(_Float16)(a3.z+b3.z); a[15]=(_Float16)(a3.w+b3.w);
  return a;
}

// ---------------- B-fragment (16-bit B 32x16 layout: lane = N col, elems = K) ----------
__device__ __forceinline__ v16h loadB(const unsigned short* __restrict__ row, int g) {
  union { unsigned short s[16]; v16h v; } u;
#pragma unroll
  for (int i = 0; i < 16; ++i) u.s[i] = row[(i + (i & 8)) * LDS_PITCH + g * 16];
  return u.v;
}

// ---------------- B slab staging: global f16 -> LDS via async DMA -----------
// One GLOBAL_LOAD_ASYNC_TO_LDS_B128 per thread: 256 threads x 16B = 4KB slab
// (32 K-rows x 64 cols f16). Tracked by ASYNCcnt, no VGPR round-trip.
__device__ __forceinline__ void stageB_async(unsigned short* dst, int ks,
                                             int l, int W,
                                             const _Float16* WT, int tid) {
  const int krel  = tid >> 3;                    // 0..31
  const int chunk = tid & 7;                     // 8 x 8-f16 chunks per row
  const int g     = chunk >> 1;                  // gate block 0..3
  const int nn    = (chunk & 1) * 8;
  const int kglob = ks * 32 + krel;
  const int ncol  = (g << 9) + (W << 4) + nn;    // 512*g + 16*W + nn
  const _Float16* src =
      WT + (size_t)l * K_ * G_ + (size_t)kglob * G_ + ncol;
  // LDS byte address = low 32 bits of the flat shared-memory pointer
  const unsigned lds = (unsigned)(uintptr_t)(dst + krel * LDS_PITCH + chunk * 8);
  asm volatile("global_load_async_to_lds_b128 %0, %1, off"
               :: "v"(lds), "v"(src)
               : "memory");
}

__device__ __forceinline__ void wait_async0() {
  asm volatile("s_wait_asynccnt 0x0" ::: "memory");
}

// ---------------- grid-wide barrier (monotone counter) ----------------------
__device__ __forceinline__ void grid_barrier(unsigned* __restrict__ bar, unsigned target) {
  __threadfence();
  __syncthreads();
  if (threadIdx.x == 0) {
    __hip_atomic_fetch_add(bar, 1u, __ATOMIC_RELEASE, __HIP_MEMORY_SCOPE_AGENT);
    while (__hip_atomic_load(bar, __ATOMIC_ACQUIRE, __HIP_MEMORY_SCOPE_AGENT) < target)
      __builtin_amdgcn_s_sleep(1);
  }
  __syncthreads();
}

// ---------------- one-time weight repack: W_ih/W_hh (f32) -> WcombT (f16) ---
__global__ void prep_weights(const float* __restrict__ Wih,
                             const float* __restrict__ Whh,
                             _Float16* __restrict__ WT) {
  const size_t idx = (size_t)blockIdx.x * blockDim.x + threadIdx.x; // 2*1024*2048
  const int l = (int)(idx >> 21);
  const int r = (int)(idx & ((1u << 21) - 1));
  const int k = r >> 11;          // 0..1023
  const int n = r & 2047;         // 0..2047
  float v;
  if (k < H_) v = Wih[((size_t)l * G_ + n) * H_ + k];
  else        v = Whh[((size_t)l * G_ + n) * H_ + (k - H_)];
  WT[idx] = (_Float16)v;
}

// ---------------- per-launch state init (rings, bias, barrier) --------------
__global__ void init_state(const float* __restrict__ h0, const float* __restrict__ c0,
                           const float* __restrict__ bih, const float* __restrict__ bhh,
                           float* __restrict__ hring, float* __restrict__ cring,
                           float* __restrict__ bias, unsigned* __restrict__ bar) {
  const int idx = blockIdx.x * blockDim.x + threadIdx.x; // 2*128*512 = 131072
  const int l = idx >> 16;
  const int r = idx & 65535;
  const size_t base = (size_t)l * 3 * B_ * H_;
  hring[base + 2 * (size_t)B_ * H_ + r] = h0[(size_t)l * B_ * H_ + r]; // slot for t=-1
  hring[base + 1 * (size_t)B_ * H_ + r] = 0.0f;                       // slot for t=-2
  cring[base + 2 * (size_t)B_ * H_ + r] = c0[(size_t)l * B_ * H_ + r];
  cring[base + 1 * (size_t)B_ * H_ + r] = 0.0f;
  if (idx < 2 * G_) bias[idx] = bih[idx] + bhh[idx];
  if (idx == 0) *bar = 0u;
}

// ---------------- persistent recurrent kernel -------------------------------
__global__ void __launch_bounds__(256, 1)
lstm_persistent(const float* __restrict__ xseq,
                const _Float16* __restrict__ WT,
                const float* __restrict__ bias,
                float* __restrict__ hring, float* __restrict__ cring,
                float* __restrict__ out, unsigned* __restrict__ bar) {
  __shared__ __align__(16) unsigned short Bs[2][32 * LDS_PITCH];

  const int tid  = threadIdx.x;
  const int wave = tid >> 5;          // 0..7 = M-tile (rows 16*wave..)
  const int lane = tid & 31;
  const int W    = blockIdx.x;        // 0..31 = h-column strip
  const int m    = lane & 15;
  const int kh   = lane >> 4;
  const int brow = (wave << 4) + m;   // A-matrix row for this lane
  unsigned bar_t = 0;

  for (int t = 0; t < T_; ++t) {
    for (int l = 0; l < 2; ++l) {
      const float* xb = (l == 0)
          ? xseq + (size_t)t * B_ * H_
          : hring + (size_t)(t % 3) * B_ * H_;                        // layer0 slot t
      const float* h1 = hring + ((size_t)l * 3 + (t + 2) % 3) * B_ * H_;  // t-1
      const float* h2 = hring + ((size_t)l * 3 + (t + 1) % 3) * B_ * H_;  // t-2

      // prefetch next timestep's input while we grind this step
      if (l == 0 && t + 1 < T_)
        __builtin_prefetch(xseq + (size_t)(t + 1) * B_ * H_ + (size_t)tid * 64, 0, 1);

      v8f acc0 = {}, acc1 = {}, acc2 = {}, acc3 = {};

      stageB_async(&Bs[0][0], 0, l, W, WT, tid);
      wait_async0();
      __syncthreads();

      for (int ks = 0; ks < NK; ++ks) {
        if (ks + 1 < NK) stageB_async(&Bs[(ks + 1) & 1][0], ks + 1, l, W, WT, tid);

        v16h a;
        if (ks < 16) {                           // x half (K 0..511)
          a = buildA1(xb + (size_t)brow * H_ + ks * 32 + kh * 8);
        } else {                                 // h half: h_{t-1}+h_{t-2}
          const int ko = (ks - 16) * 32 + kh * 8;
          a = buildA2(h1 + (size_t)brow * H_ + ko, h2 + (size_t)brow * H_ + ko);
        }

        const unsigned short* bsrow = &Bs[ks & 1][(kh * 8) * LDS_PITCH + m];
        acc0 = wmma16(a, loadB(bsrow, 0), acc0); // i
        acc1 = wmma16(a, loadB(bsrow, 1), acc1); // f
        acc2 = wmma16(a, loadB(bsrow, 2), acc2); // g
        acc3 = wmma16(a, loadB(bsrow, 3), acc3); // o

        wait_async0();    // async slab for ks+1 has landed (hidden by WMMAs)
        __syncthreads();
      }

      // ---- LSTM cell, fully in-register (i,f,g,o share (vgpr j, lane)) ----
      {
        const int n   = (W << 4) + (lane & 15);
        const float bi = bias[l * G_ + n];
        const float bf = bias[l * G_ + 512 + n];
        const float bg = bias[l * G_ + 1024 + n];
        const float bo = bias[l * G_ + 1536 + n];
        float* ho = hring + ((size_t)l * 3 + t % 3) * B_ * H_;
        float* co = cring + ((size_t)l * 3 + t % 3) * B_ * H_;
        const float* c1 = cring + ((size_t)l * 3 + (t + 2) % 3) * B_ * H_;
        const float* c2 = cring + ((size_t)l * 3 + (t + 1) % 3) * B_ * H_;
#pragma unroll
        for (int j = 0; j < 8; ++j) {
          const int b = (wave << 4) + j + ((lane >> 4) << 3);  // C/D layout row
          const size_t o = (size_t)b * H_ + n;
          const float iv = fsig(acc0[j] + bi);
          const float fv = fsig(acc1[j] + bf);
          const float gv = ftanh(acc2[j] + bg);
          const float ov = fsig(acc3[j] + bo);
          const float cn = fv * (c1[o] + c2[o]) + iv * gv;
          const float hn = ov * ftanh(cn);
          co[o] = cn;
          ho[o] = hn;
          if (l == 1 && t == T_ - 1) out[o] = hn;
        }
      }

      bar_t += NWG;
      grid_barrier(bar, bar_t);
    }
  }
}

// ---------------------------------------------------------------------------
extern "C" void kernel_launch(void* const* d_in, const int* in_sizes, int n_in,
                              void* d_out, int out_size, void* d_ws, size_t ws_size,
                              hipStream_t stream) {
  const float* xseq = (const float*)d_in[0];   // [T,B,H]
  const float* h0   = (const float*)d_in[1];   // [L,B,H]
  const float* c0   = (const float*)d_in[2];   // [L,B,H]
  const float* Wih  = (const float*)d_in[3];   // [L,4H,H]
  const float* Whh  = (const float*)d_in[4];   // [L,4H,H]
  const float* bih  = (const float*)d_in[5];   // [L,4H]
  const float* bhh  = (const float*)d_in[6];   // [L,4H]
  float* out = (float*)d_out;                  // [B,H]

  // workspace layout (~11.6 MB total)
  char* ws = (char*)d_ws;
  unsigned*  bar   = (unsigned*)ws;                          // 256 B
  float*     bias  = (float*)(ws + 256);                     // 2*2048*4   = 16 KB
  _Float16*  WT    = (_Float16*)(ws + 256 + 16384);          // 2*1024*2048*2 = 8 MB
  float*     hring = (float*)(ws + 256 + 16384 + 8388608);   // 2*3*128*512*4 = 1.5 MB
  float*     cring = hring + (size_t)2 * 3 * B_ * H_;        // 1.5 MB

  prep_weights<<<(2 * K_ * G_) / 256, 256, 0, stream>>>(Wih, Whh, WT);
  init_state<<<(2 * B_ * H_) / 256, 256, 0, stream>>>(h0, c0, bih, bhh,
                                                      hring, cring, bias, bar);
  lstm_persistent<<<NWG, 256, 0, stream>>>(xseq, WT, bias, hring, cring, out, bar);
}
